// PointwiseAggregatedAttention_86955907875119
// MI455X (gfx1250) — matrix operand order repaired and checked by reading, not compiled, not run on hardware
//
#include <hip/hip_runtime.h>
#include <hip/hip_bf16.h>

#define BB  4
#define SS  2048
#define DD  256
#define HH  4
#define DHH 64

typedef __attribute__((ext_vector_type(16))) _Float16     v16h;
typedef __attribute__((ext_vector_type(8)))  _Float16     v8h;
typedef __attribute__((ext_vector_type(8)))  float        v8f;
typedef __attribute__((ext_vector_type(4)))  unsigned int u32x4;
typedef __attribute__((ext_vector_type(8)))  int          i32x8;
typedef __attribute__((ext_vector_type(4)))  int          i32x4;

// ---------------- WMMA helpers (CDNA5 wave32, 16x16x32 f16 -> f32) ----------

static __device__ __forceinline__ v8f wmma_f16(v16h a, v16h b, v8f c) {
  // (neg_a, A, neg_b, B, c_mod, C, reuse_a, reuse_b)
  return __builtin_amdgcn_wmma_f32_16x16x32_f16(false, a, false, b, (short)0, c,
                                                false, false);
}

// A fragment: 16x32 f16, row-major source with row stride ld (elements).
static __device__ __forceinline__ v16h load_a_frag(const _Float16* src, int ld,
                                                   int lane) {
  const int m  = lane & 15;
  const int hi = (lane >> 4) & 1;
  const _Float16* row = src + m * ld + hi * 8;
  v8h lo = *(const v8h*)(row);
  v8h hh = *(const v8h*)(row + 16);
  return __builtin_shufflevector(lo, hh, 0, 1, 2, 3, 4, 5, 6, 7, 8, 9, 10, 11,
                                 12, 13, 14, 15);
}

// B fragment: B[k][n] = src[n*ld + k] (transposed read, two 16B runs/lane).
static __device__ __forceinline__ v16h load_bT_frag(const _Float16* src, int ld,
                                                    int lane) {
  const int n   = lane & 15;
  const int klo = (lane & 16) >> 1;
  const _Float16* col = src + n * ld + klo;
  v8h lo = *(const v8h*)(col);
  v8h hh = *(const v8h*)(col + 16);
  return __builtin_shufflevector(lo, hh, 0, 1, 2, 3, 4, 5, 6, 7, 8, 9, 10, 11,
                                 12, 13, 14, 15);
}

// ---------------- Tensor Data Mover staging ----------------

static __device__ __forceinline__ unsigned lds_off(const void* p) {
  return (unsigned)(unsigned long long)p;  // low 32 bits of generic = LDS offset
}

// Async DMA of a 2D f16 tile (w elements wide, h rows, row stride in elements)
// from global to LDS (row-major, packed w*2 bytes per row). Tracked: TENSORcnt.
static __device__ __forceinline__ void tdm_load_2d(const _Float16* gptr,
                                                   unsigned ldsoff, unsigned w,
                                                   unsigned h, unsigned stride) {
#if __has_builtin(__builtin_amdgcn_tensor_load_to_lds)
  unsigned long long ga = (unsigned long long)gptr;
  u32x4 g0;
  g0[0] = 1u;                                         // count=1, user descriptor
  g0[1] = ldsoff;                                     // lds_addr
  g0[2] = (unsigned)(ga & 0xffffffffu);               // global_addr[31:0]
  g0[3] = (unsigned)((ga >> 32) & 0x1ffffffu) | (2u << 30);  // addr[56:32],type=2
  i32x8 g1;
  g1[0] = (int)(1u << 16);   // workgroup_mask=0, data_size=1 (2 bytes), no flags
  g1[1] = (int)(w << 16);    // tensor_dim0[15:0]  (bits 79:48)
  g1[2] = (int)(h << 16);    // tensor_dim1[15:0]  (bits 111:80)
  g1[3] = (int)(w << 16);    // tile_dim0          (bits 127:112)
  g1[4] = (int)h;            // tile_dim1          (bits 143:128), tile_dim2=0
  g1[5] = (int)stride;       // tensor_dim0_stride[31:0]
  g1[6] = 0;
  g1[7] = 0;
  i32x4 z4 = {0, 0, 0, 0};
#if __clang_major__ >= 23
  i32x8 z8 = {0, 0, 0, 0, 0, 0, 0, 0};
  __builtin_amdgcn_tensor_load_to_lds(g0, g1, z4, z4, z8, 0);
#else
  __builtin_amdgcn_tensor_load_to_lds(g0, g1, z4, z4, 0);
#endif
#else
  (void)gptr; (void)ldsoff; (void)w; (void)h; (void)stride;
#endif
}

template <int N>
static __device__ __forceinline__ void wait_tensorcnt() {
#if __has_builtin(__builtin_amdgcn_s_wait_tensorcnt)
  __builtin_amdgcn_s_wait_tensorcnt(N);  // SIMM16 must be a literal constant
#endif
}

// ---------------- precision conversion ----------------

__global__ void cvt_f32_f16(const float* __restrict__ src,
                            _Float16* __restrict__ dst, int n) {
  int i = blockIdx.x * blockDim.x + threadIdx.x;
  if (i < n) dst[i] = (_Float16)src[i];
}

// ---------------- QKV projection: Y = x @ W^T + b, split heads -------------
// One wave per 16x64 output tile. z selects Q/K/V. 1/scale folded into Q.
// V is stored TRANSPOSED per head: VhT[b][h][dh][s]  (for clean attention B-frags).

__global__ __launch_bounds__(32) void qkv_proj(
    const _Float16* __restrict__ xh, const _Float16* __restrict__ Wqh,
    const _Float16* __restrict__ Wkh, const _Float16* __restrict__ Wvh,
    const float* __restrict__ bq, const float* __restrict__ bk,
    const float* __restrict__ bv, const float* __restrict__ scale,
    _Float16* __restrict__ Qh, _Float16* __restrict__ Kh,
    _Float16* __restrict__ VhT) {
  const int lane = threadIdx.x & 31;
  const int mt = blockIdx.x;       // 16-row tile of flattened (B*S)
  const int jb = blockIdx.y * 64;  // output-column base
  const int z  = blockIdx.z;

  const _Float16* W    = (z == 0) ? Wqh : (z == 1) ? Wkh : Wvh;
  const float*    bias = (z == 0) ? bq  : (z == 1) ? bk  : bv;
  _Float16*       Out  = (z == 0) ? Qh  : (z == 1) ? Kh  : VhT;
  const float     mul  = (z == 0) ? (1.0f / scale[0]) : 1.0f;

  v8f acc[4] = {};
  const _Float16* Arow = xh + (size_t)mt * 16 * DD;

#pragma unroll
  for (int kc = 0; kc < DD / 32; ++kc) {
    v16h a = load_a_frag(Arow + kc * 32, DD, lane);
#pragma unroll
    for (int nb = 0; nb < 4; ++nb) {
      v16h b = load_bT_frag(W + (size_t)(jb + nb * 16) * DD + kc * 32, DD, lane);
      acc[nb] = wmma_f16(a, b, acc[nb]);
    }
  }

  const int hi8 = (lane & 16) ? 8 : 0;
  const int n   = lane & 15;
#pragma unroll
  for (int nb = 0; nb < 4; ++nb) {
    const int   j  = jb + nb * 16 + n;
    const float bj = bias[j];
    const int   h  = j >> 6;  // DH = 64
    const int   dh = j & 63;
#pragma unroll
    for (int r = 0; r < 8; ++r) {
      const int t    = mt * 16 + r + hi8;  // flattened b*S + s
      const int bidx = t >> 11;
      const int sidx = t & (SS - 1);
      float v = (acc[nb][r] + bj) * mul;
      if (z == 2)  // V transposed: [b][h][dh][s]
        Out[((size_t)(bidx * HH + h) * DHH + dh) * SS + sidx] = (_Float16)v;
      else         // Q/K: [b][h][s][dh]
        Out[((size_t)(bidx * HH + h) * SS + sidx) * DHH + dh] = (_Float16)v;
    }
  }
}

// ---------------- fused attention: O = (Q K^T * 1/scale + bias, mask) V ----
// 4 waves/block, each wave a 16-row Q tile. K (32x64) and V^T (64x32) chunks
// are DMA'd into double-buffered LDS by the Tensor Data Mover; scores never
// touch HBM.

__global__ __launch_bounds__(128) void attn(
    const _Float16* __restrict__ Qh, const _Float16* __restrict__ Kh,
    const _Float16* __restrict__ VhT, const float* __restrict__ rel_bias,
    const int* __restrict__ mask, _Float16* __restrict__ Oh) {
  __shared__ __align__(32) _Float16 Kld[2][32 * DHH];   // [k][dh]
  __shared__ __align__(32) _Float16 Vld[2][DHH * 32];   // [dh][k] (transposed)
  __shared__ __align__(32) _Float16 Sld[4][16 * 32];
  __shared__ float Rb[2 * SS - 1];
  __shared__ int   Mld[SS];

  const int tid  = threadIdx.x;
  const int lane = tid & 31;
  const int wv   = tid >> 5;
  const int b = blockIdx.z, h = blockIdx.y;
  const int qrow0 = blockIdx.x * 64 + wv * 16;

  const _Float16* Qb  = Qh  + (size_t)(b * HH + h) * SS * DHH;
  const _Float16* Kb  = Kh  + (size_t)(b * HH + h) * SS * DHH;
  const _Float16* VbT = VhT + (size_t)(b * HH + h) * DHH * SS;

  // Kick off DMA of chunk 0 (wave 0 only; TDM is a wave-level op).
  if (wv == 0) {
    tdm_load_2d(Kb, lds_off(&Kld[0][0]), 64, 32, 64);    // K rows 0..31
    tdm_load_2d(VbT, lds_off(&Vld[0][0]), 32, DHH, SS);  // V^T cols 0..31
  }

  // one-time LDS staging: rel-bias column h, mask row b
  for (int t = tid; t < 2 * SS - 1; t += 128) Rb[t] = rel_bias[(size_t)t * HH + h];
  for (int t = tid; t < SS; t += 128) Mld[t] = mask[b * SS + t];
  __syncthreads();

  // Q fragments, 16x64 = two 16x32 A frags (scale already folded in)
  v16h qa0 = load_a_frag(Qb + (size_t)qrow0 * DHH, DHH, lane);
  v16h qa1 = load_a_frag(Qb + (size_t)qrow0 * DHH + 32, DHH, lane);

  const int hi8 = (lane & 16) ? 8 : 0;
  const int n   = lane & 15;
  unsigned qmask = 0;
#pragma unroll
  for (int r = 0; r < 8; ++r)
    if (Mld[qrow0 + r + hi8] != 0) qmask |= 1u << r;

  v8f acc[4] = {};
  _Float16* Ss = &Sld[wv][0];

  for (int kc = 0; kc < SS / 32; ++kc) {
    const int buf = kc & 1;
    if (wv == 0) {
      if (kc + 1 < SS / 32) {  // prefetch next chunk into the other buffer
        tdm_load_2d(Kb + (size_t)(kc + 1) * 32 * DHH, lds_off(&Kld[buf ^ 1][0]),
                    64, 32, 64);
        tdm_load_2d(VbT + (size_t)(kc + 1) * 32, lds_off(&Vld[buf ^ 1][0]),
                    32, DHH, SS);
        wait_tensorcnt<2>();  // oldest two DMAs (current chunk) have landed
      } else {
        wait_tensorcnt<0>();  // final chunk: drain everything
      }
    }
    __syncthreads();

    const _Float16* Kc = &Kld[buf][0];
    const _Float16* Vc = &Vld[buf][0];

    // scores: two 16x16 tiles (contraction over DH=64 in two K=32 steps)
#pragma unroll
    for (int jt = 0; jt < 2; ++jt) {
      v8f c = {};
      c = wmma_f16(qa0, load_bT_frag(Kc + jt * 16 * DHH, DHH, lane), c);
      c = wmma_f16(qa1, load_bT_frag(Kc + jt * 16 * DHH + 32, DHH, lane), c);
      // rel-bias + pairwise mask, convert to f16, park in per-wave LDS scratch
#pragma unroll
      for (int r = 0; r < 8; ++r) {
        const int i  = qrow0 + r + hi8;
        const int jg = kc * 32 + jt * 16 + n;
        float v;
        if (((qmask >> r) & 1u) && Mld[jg])
          v = c[r] + Rb[jg - i + (SS - 1)];
        else
          v = -__builtin_inff();
        Ss[(r + hi8) * 32 + jt * 16 + n] = (_Float16)v;
      }
    }

    // O += scores(16x32) @ V(32x64); V^T in LDS makes every B-frag contiguous
    v16h sa = load_a_frag(Ss, 32, lane);
#pragma unroll
    for (int nb = 0; nb < 4; ++nb) {
      v16h bvf = load_bT_frag(Vc + nb * 16 * 32, 32, lane);
      acc[nb] = wmma_f16(sa, bvf, acc[nb]);
    }
    __syncthreads();  // all waves done with buf before its DMA overwrite
  }

  // write O as [b][s][h*DH+dh] (== (B,S,D) after head merge)
#pragma unroll
  for (int nb = 0; nb < 4; ++nb) {
#pragma unroll
    for (int r = 0; r < 8; ++r) {
      const int s = qrow0 + r + hi8;
      Oh[(size_t)(b * SS + s) * DD + h * DHH + nb * 16 + n] =
          (_Float16)acc[nb][r];
    }
  }
}

// ---------------- output projection: out = O @ Wo^T + bo (f32 out) ---------

__global__ __launch_bounds__(32) void out_proj(
    const _Float16* __restrict__ Oh, const _Float16* __restrict__ Woh,
    const float* __restrict__ bo, float* __restrict__ out) {
  const int lane = threadIdx.x & 31;
  const int mt = blockIdx.x;
  const int jb = blockIdx.y * 64;

  v8f acc[4] = {};
  const _Float16* Arow = Oh + (size_t)mt * 16 * DD;

#pragma unroll
  for (int kc = 0; kc < DD / 32; ++kc) {
    v16h a = load_a_frag(Arow + kc * 32, DD, lane);
#pragma unroll
    for (int nb = 0; nb < 4; ++nb) {
      v16h b =
          load_bT_frag(Woh + (size_t)(jb + nb * 16) * DD + kc * 32, DD, lane);
      acc[nb] = wmma_f16(a, b, acc[nb]);
    }
  }

  const int hi8 = (lane & 16) ? 8 : 0;
  const int n   = lane & 15;
#pragma unroll
  for (int nb = 0; nb < 4; ++nb) {
    const int   j  = jb + nb * 16 + n;
    const float bj = bo[j];
#pragma unroll
    for (int r = 0; r < 8; ++r) {
      const int t = mt * 16 + r + hi8;
      out[(size_t)t * DD + j] = acc[nb][r] + bj;
    }
  }
}

// ---------------- host launch ----------------

extern "C" void kernel_launch(void* const* d_in, const int* in_sizes, int n_in,
                              void* d_out, int out_size, void* d_ws,
                              size_t ws_size, hipStream_t stream) {
  (void)in_sizes; (void)n_in; (void)out_size; (void)ws_size;

  const float* x    = (const float*)d_in[0];
  const float* Wq   = (const float*)d_in[1];
  const float* bq   = (const float*)d_in[2];
  const float* Wk   = (const float*)d_in[3];
  const float* bk   = (const float*)d_in[4];
  const float* Wv   = (const float*)d_in[5];
  const float* bv   = (const float*)d_in[6];
  const float* Wo   = (const float*)d_in[7];
  const float* bo   = (const float*)d_in[8];
  const float* rb   = (const float*)d_in[9];
  const float* sc   = (const float*)d_in[10];
  const int*   mask = (const int*)d_in[11];
  float*       out  = (float*)d_out;

  char* ws = (char*)d_ws;
  _Float16* xh  = (_Float16*)(ws);               // 4 MB
  _Float16* Wqh = (_Float16*)(ws + (4u << 20));  // 128 KB each
  _Float16* Wkh = Wqh + 65536;
  _Float16* Wvh = Wkh + 65536;
  _Float16* Woh = Wvh + 65536;
  _Float16* Qh  = (_Float16*)(ws + (5u << 20));  // 4 MB each
  _Float16* Kh  = Qh + 2097152;
  _Float16* VhT = Kh + 2097152;
  _Float16* Oh  = VhT + 2097152;                 // total 21 MB

  const int nx = BB * SS * DD;  // 2,097,152
  const int nw = DD * DD;       // 65,536
  cvt_f32_f16<<<(nx + 255) / 256, 256, 0, stream>>>(x, xh, nx);
  cvt_f32_f16<<<(nw + 255) / 256, 256, 0, stream>>>(Wq, Wqh, nw);
  cvt_f32_f16<<<(nw + 255) / 256, 256, 0, stream>>>(Wk, Wkh, nw);
  cvt_f32_f16<<<(nw + 255) / 256, 256, 0, stream>>>(Wv, Wvh, nw);
  cvt_f32_f16<<<(nw + 255) / 256, 256, 0, stream>>>(Wo, Woh, nw);

  qkv_proj<<<dim3(BB * SS / 16, DD / 64, 3), 32, 0, stream>>>(
      xh, Wqh, Wkh, Wvh, bq, bk, bv, sc, Qh, Kh, VhT);

  attn<<<dim3(SS / 64, HH, BB), 128, 0, stream>>>(Qh, Kh, VhT, rb, mask, Oh);

  out_proj<<<dim3(BB * SS / 16, DD / 64), 32, 0, stream>>>(Oh, Woh, bo, out);
}
